// MobileNetV2ME_28140625723924
// MI455X (gfx1250) — matrix operand order repaired and checked by reading, not compiled, not run on hardware
//
#include <hip/hip_runtime.h>

typedef __attribute__((ext_vector_type(16))) _Float16 v16h;
typedef __attribute__((ext_vector_type(8)))  _Float16 h8v;
typedef __attribute__((ext_vector_type(8)))  float    v8f;

// ---------------------------------------------------------------------------
// WMMA GEMM + fused BN (+ReLU6) * mask (+residual), f16 in / f32 acc / f16 out
//   A   : [M, Kpad]   f16 row-major activations (pad cols zeroed)
//   W   : [Kpad, Npad] f16 row-major weights   (pads zeroed)
//   out : [M, Npad]   f16 ; cols >= Nreal written as 0
// One wave computes a 32x32 output macro-tile (2 M-tiles x 2 N-tiles,
// 4 f32 accumulators). Per 32-wide K step: 4 A-loads + 4 B-loads feed
// 4 v_wmma_f32_16x16x32_f16 -> 1 b128 A-load per WMMA; B gathered by strided f16 loads.
// ---------------------------------------------------------------------------
__global__ void __launch_bounds__(256)
k_gemm_bn(const _Float16* __restrict__ A, const _Float16* __restrict__ W,
          const float* __restrict__ gamma, const float* __restrict__ beta,
          const float* __restrict__ mean,  const float* __restrict__ var,
          const float* __restrict__ mask,  const _Float16* __restrict__ resid,
          _Float16* __restrict__ out,
          int M, int Kpad, int Npad, int Nreal, int relu6)
{
  const int nblk = Npad >> 5;              // 32-col macro tiles
  const int mblk = M >> 5;                 // 32-row macro tiles
  int wave = blockIdx.x * (blockDim.x >> 5) + (threadIdx.x >> 5);
  if (wave >= mblk * nblk) return;         // uniform per wave
  int mt = wave / nblk;
  int nt = wave - mt * nblk;
  int lane = threadIdx.x & 31;
  int grp  = lane >> 4;
  int hl   = lane & 15;

  // A fragments: lane holds row M = hl (+16), K chunks grp*8.. / 16+grp*8..
  const _Float16* arow0 = A + (size_t)(mt * 32 + hl) * Kpad;
  const _Float16* arow1 = arow0 + (size_t)16 * Kpad;
  // B fragments: lane holds column N = hl (+16), K chunks grp*8.. / 16+grp*8..
  const int bk = hl + 16 * grp;
  const _Float16* wbase = W + (size_t)nt * 32;

  v8f acc00 = {}, acc01 = {}, acc10 = {}, acc11 = {};
  for (int k0 = 0; k0 < Kpad; k0 += 32) {
    h8v a0lo = *(const h8v*)(arow0 + k0 + grp * 8);
    h8v a0hi = *(const h8v*)(arow0 + k0 + 16 + grp * 8);
    h8v a1lo = *(const h8v*)(arow1 + k0 + grp * 8);
    h8v a1hi = *(const h8v*)(arow1 + k0 + 16 + grp * 8);
    const _Float16* wr = wbase + (size_t)k0 * Npad + hl;
    h8v b0lo, b0hi, b1lo, b1hi;
#pragma unroll
    for (int i = 0; i < 8; ++i) {
      b0lo[i] = wr[(size_t)(grp * 8 + i) * Npad];
      b0hi[i] = wr[(size_t)(16 + grp * 8 + i) * Npad];
      b1lo[i] = wr[(size_t)(grp * 8 + i) * Npad + 16];
      b1hi[i] = wr[(size_t)(16 + grp * 8 + i) * Npad + 16];
    }
    v16h a0, a1, b0, b1;
#pragma unroll
    for (int i = 0; i < 8; ++i) {
      a0[i] = a0lo[i]; a0[i+8] = a0hi[i];
      a1[i] = a1lo[i]; a1[i+8] = a1hi[i];
      b0[i] = b0lo[i]; b0[i+8] = b0hi[i];
      b1[i] = b1lo[i]; b1[i+8] = b1hi[i];
    }
    acc00 = __builtin_amdgcn_wmma_f32_16x16x32_f16(false, a0, false, b0, (short)0, acc00, false, false);
    acc01 = __builtin_amdgcn_wmma_f32_16x16x32_f16(false, a0, false, b1, (short)0, acc01, false, false);
    acc10 = __builtin_amdgcn_wmma_f32_16x16x32_f16(false, a1, false, b0, (short)0, acc10, false, false);
    acc11 = __builtin_amdgcn_wmma_f32_16x16x32_f16(false, a1, false, b1, (short)0, acc11, false, false);
  }

  // Epilogue. C layout: VGPR i, lane -> (row = i + 8*grp, col = hl)
  int n0 = nt * 32 + hl;
  int n1 = n0 + 16;
  int n0c = (n0 < Nreal) ? n0 : (Nreal - 1);
  int n1c = (n1 < Nreal) ? n1 : (Nreal - 1);
  float g0 = gamma[n0c], t0 = beta[n0c], u0 = mean[n0c];
  float r0 = rsqrtf(var[n0c] + 1e-5f);
  float g1 = gamma[n1c], t1 = beta[n1c], u1 = mean[n1c];
  float r1 = rsqrtf(var[n1c] + 1e-5f);
#pragma unroll
  for (int i = 0; i < 8; ++i) {
    int rowA = mt * 32 + i + 8 * grp;
    int rowB = rowA + 16;
    float mA = mask[rowA];
    float mB = mask[rowB];
    float v00 = (g0 * (acc00[i] - u0) * r0 + t0);
    float v01 = (g1 * (acc01[i] - u1) * r1 + t1);
    float v10 = (g0 * (acc10[i] - u0) * r0 + t0);
    float v11 = (g1 * (acc11[i] - u1) * r1 + t1);
    if (relu6) {
      v00 = fminf(fmaxf(v00, 0.f), 6.f);
      v01 = fminf(fmaxf(v01, 0.f), 6.f);
      v10 = fminf(fmaxf(v10, 0.f), 6.f);
      v11 = fminf(fmaxf(v11, 0.f), 6.f);
    }
    v00 *= mA; v01 *= mA; v10 *= mB; v11 *= mB;
    if (resid) {
      v00 += (float)resid[(size_t)rowA * Npad + n0];
      v01 += (float)resid[(size_t)rowA * Npad + n1];
      v10 += (float)resid[(size_t)rowB * Npad + n0];
      v11 += (float)resid[(size_t)rowB * Npad + n1];
    }
    out[(size_t)rowA * Npad + n0] = (n0 < Nreal) ? (_Float16)v00 : (_Float16)0.f;
    out[(size_t)rowA * Npad + n1] = (n1 < Nreal) ? (_Float16)v01 : (_Float16)0.f;
    out[(size_t)rowB * Npad + n0] = (n0 < Nreal) ? (_Float16)v10 : (_Float16)0.f;
    out[(size_t)rowB * Npad + n1] = (n1 < Nreal) ? (_Float16)v11 : (_Float16)0.f;
  }
}

// ---------------------------------------------------------------------------
// Depthwise 3x3 conv (SAME) + BN + ReLU6 * mask ; f16 act, f32 weights
// ---------------------------------------------------------------------------
__global__ void k_dwconv_bn(const _Float16* __restrict__ in, const float* __restrict__ w,
                            const float* __restrict__ g,  const float* __restrict__ bt,
                            const float* __restrict__ mu, const float* __restrict__ var,
                            const float* __restrict__ mout, _Float16* __restrict__ out,
                            int B, int Hin, int Win, int Hout, int Wout,
                            int C, int Cpad, int stride, int padlo)
{
  size_t i = (size_t)blockIdx.x * blockDim.x + threadIdx.x;
  size_t total = (size_t)B * Hout * Wout * Cpad;
  if (i >= total) return;
  int c = (int)(i % Cpad);
  size_t s = i / Cpad;
  int xo = (int)(s % Wout); s /= Wout;
  int yo = (int)(s % Hout);
  int b  = (int)(s / Hout);
  if (c >= C) { out[i] = (_Float16)0.f; return; }
  float acc = 0.f;
#pragma unroll
  for (int ky = 0; ky < 3; ++ky) {
    int yi = yo * stride + ky - padlo;
    if (yi < 0 || yi >= Hin) continue;
#pragma unroll
    for (int kx = 0; kx < 3; ++kx) {
      int xi = xo * stride + kx - padlo;
      if (xi < 0 || xi >= Win) continue;
      float hv = (float)in[(((size_t)b * Hin + yi) * Win + xi) * Cpad + c];
      acc += hv * w[(ky * 3 + kx) * C + c];
    }
  }
  float v = g[c] * (acc - mu[c]) * rsqrtf(var[c] + 1e-5f) + bt[c];
  v = fminf(fmaxf(v, 0.f), 6.f);
  v *= mout[((size_t)b * Hout + yo) * Wout + xo];
  out[i] = (_Float16)v;
}

// ---------------------------------------------------------------------------
// Stem: 3x3 s2 SAME conv on (x*mask), 2 -> 32 ch, + BN + ReLU6 * m1, f16 out
// ---------------------------------------------------------------------------
__global__ void k_stem(const float* __restrict__ x, const float* __restrict__ m0,
                       const float* __restrict__ w,
                       const float* __restrict__ g,  const float* __restrict__ bt,
                       const float* __restrict__ mu, const float* __restrict__ var,
                       const float* __restrict__ m1, _Float16* __restrict__ out, int B)
{
  size_t i = (size_t)blockIdx.x * blockDim.x + threadIdx.x;
  size_t total = (size_t)B * 128 * 128 * 32;
  if (i >= total) return;
  int co = (int)(i % 32);
  size_t s = i / 32;
  int xo = (int)(s % 128); s /= 128;
  int yo = (int)(s % 128);
  int b  = (int)(s / 128);
  float acc = 0.f;
#pragma unroll
  for (int ky = 0; ky < 3; ++ky) {
    int yi = 2 * yo + ky;                   // SAME s2: pad_lo = 0
    if (yi >= 256) continue;
#pragma unroll
    for (int kx = 0; kx < 3; ++kx) {
      int xi = 2 * xo + kx;
      if (xi >= 256) continue;
      size_t sp = ((size_t)b * 256 + yi) * 256 + xi;
      float mk = m0[sp];
      const float* xp = x + sp * 2;
      acc += mk * (xp[0] * w[((ky * 3 + kx) * 2 + 0) * 32 + co] +
                   xp[1] * w[((ky * 3 + kx) * 2 + 1) * 32 + co]);
    }
  }
  float v = g[co] * (acc - mu[co]) * rsqrtf(var[co] + 1e-5f) + bt[co];
  v = fminf(fmaxf(v, 0.f), 6.f);
  v *= m1[((size_t)b * 128 + yo) * 128 + xo];
  out[i] = (_Float16)v;
}

// 3x3 stride-2 SAME max-pool on the occupancy mask
__global__ void k_downmask(const float* __restrict__ mi, float* __restrict__ mo,
                           int B, int Hin, int Win, int Hout, int Wout)
{
  size_t i = (size_t)blockIdx.x * blockDim.x + threadIdx.x;
  size_t total = (size_t)B * Hout * Wout;
  if (i >= total) return;
  int xo = (int)(i % Wout);
  size_t s = i / Wout;
  int yo = (int)(s % Hout);
  int b  = (int)(s / Hout);
  float mx = 0.f;
#pragma unroll
  for (int ky = 0; ky < 3; ++ky) {
    int yi = 2 * yo + ky;
    if (yi >= Hin) continue;
#pragma unroll
    for (int kx = 0; kx < 3; ++kx) {
      int xi = 2 * xo + kx;
      if (xi >= Win) continue;
      mx = fmaxf(mx, mi[((size_t)b * Hin + yi) * Win + xi]);
    }
  }
  mo[i] = mx;
}

// fp32 [K,N] weight -> padded f16 [Kpad,Npad] (zeros in pads)
__global__ void k_prep_w(const float* __restrict__ w, _Float16* __restrict__ o,
                         int K, int N, int Kpad, int Npad)
{
  int i = blockIdx.x * blockDim.x + threadIdx.x;
  if (i >= Kpad * Npad) return;
  int k = i / Npad, n = i % Npad;
  o[i] = (_Float16)((k < K && n < N) ? w[k * N + n] : 0.f);
}

// masked global average pool over S active-weighted sites
__global__ void k_pool(const _Float16* __restrict__ h, const float* __restrict__ m,
                       float* __restrict__ pooled, int B, int S, int C)
{
  int i = blockIdx.x * blockDim.x + threadIdx.x;
  if (i >= B * C) return;
  int b = i / C, c = i % C;
  float num = 0.f, den = 0.f;
  for (int s = 0; s < S; ++s) {
    float mv = m[b * S + s];
    num += (float)h[((size_t)(b * S + s)) * C + c] * mv;
    den += mv;
  }
  pooled[i] = num / fmaxf(den, 1.f);
}

__global__ void k_fc(const float* __restrict__ p, const float* __restrict__ w,
                     const float* __restrict__ bias, float* __restrict__ out,
                     int B, int C, int NC)
{
  int i = blockIdx.x * blockDim.x + threadIdx.x;
  if (i >= B * NC) return;
  int b = i / NC, j = i % NC;
  float acc = bias[j];
  for (int c = 0; c < C; ++c) acc += p[b * C + c] * w[c * NC + j];
  out[i] = acc;
}

// ---------------------------------------------------------------------------
static inline int pad32(int x) { return (x + 31) & ~31; }

extern "C" void kernel_launch(void* const* d_in, const int* in_sizes, int n_in,
                              void* d_out, int out_size, void* d_ws, size_t ws_size,
                              hipStream_t stream)
{
  (void)in_sizes; (void)n_in; (void)out_size; (void)ws_size;
  const float* x  = (const float*)d_in[0];
  const float* m0 = (const float*)d_in[1];
  // params: JAX pytree leaf order (dict keys sorted alphabetically)
  auto P = [&](int i) { return (const float*)d_in[2 + i]; };
  const int NB = 17;
  // leaves: blocks[b]: bn1(g,b,m,v) bn2(..) bn3(..) w1 w2 w3  => 15 leaves/block
  const float* fc_b = P(NB * 15 + 0);
  const float* fc_w = P(NB * 15 + 1);
  const float* head_bn[4]; for (int i = 0; i < 4; ++i) head_bn[i] = P(NB * 15 + 2 + i);
  const float* head_w = P(NB * 15 + 6);
  const float* stem_bn[4]; for (int i = 0; i < 4; ++i) stem_bn[i] = P(NB * 15 + 7 + i);
  const float* stem_w = P(NB * 15 + 11);

  struct Blk { int cin, hid, cout, stride, Hin; };
  const Blk blk[NB] = {
    {32, 32, 16, 1, 128},
    {16, 96, 24, 2, 128}, {24, 144, 24, 1, 64},
    {24, 144, 32, 2, 64}, {32, 192, 32, 1, 32}, {32, 192, 32, 1, 32},
    {32, 192, 64, 2, 32}, {64, 384, 64, 1, 16}, {64, 384, 64, 1, 16}, {64, 384, 64, 1, 16},
    {64, 384, 96, 1, 16}, {96, 576, 96, 1, 16}, {96, 576, 96, 1, 16},
    {96, 576, 160, 2, 16}, {160, 960, 160, 1, 8}, {160, 960, 160, 1, 8},
    {160, 960, 320, 1, 8},
  };
  const int B = 32;

  // ---- workspace carve (deterministic) ----
  char* ws = (char*)d_ws;
  size_t off = 0;
  auto carve = [&](size_t bytes) -> void* {
    off = (off + 255) & ~(size_t)255;
    void* p = ws + off; off += bytes; return p;
  };
  float* m1 = (float*)carve((size_t)B * 128 * 128 * 4);
  float* m2 = (float*)carve((size_t)B * 64 * 64 * 4);
  float* m3 = (float*)carve((size_t)B * 32 * 32 * 4);
  float* m4 = (float*)carve((size_t)B * 16 * 16 * 4);
  float* m5 = (float*)carve((size_t)B * 8 * 8 * 4);
  const size_t ACT_AB = (size_t)B * 128 * 128 * 96 * sizeof(_Float16);  // ~100.7 MB
  const size_t ACT_C  = (size_t)B * 64 * 64 * 160 * sizeof(_Float16);   // ~41.9 MB
  _Float16* actA = (_Float16*)carve(ACT_AB);
  _Float16* actB = (_Float16*)carve(ACT_AB);
  _Float16* actC = (_Float16*)carve(ACT_C);
  float* pooled  = (float*)carve((size_t)B * 1280 * 4);
  _Float16* wpool = (_Float16*)carve((size_t)8 << 20);

  // ---- weight prep: fp32 -> padded f16 ----
  _Float16* wptr = wpool;
  _Float16* w1h[NB]; _Float16* w3h[NB]; _Float16* headh;
  auto prep = [&](const float* src, int K, int N) -> _Float16* {
    int Kp = pad32(K), Np = pad32(N);
    _Float16* dst = wptr; wptr += (size_t)Kp * Np;
    int tot = Kp * Np;
    k_prep_w<<<(tot + 255) / 256, 256, 0, stream>>>(src, dst, K, N, Kp, Np);
    return dst;
  };
  for (int b = 0; b < NB; ++b) {
    w1h[b] = prep(P(b * 15 + 12), blk[b].cin, blk[b].hid);
    w3h[b] = prep(P(b * 15 + 14), blk[b].hid, blk[b].cout);
  }
  headh = prep(head_w, 320, 1280);

  // ---- mask pyramid ----
  auto down = [&](const float* mi, float* mo, int Hin) {
    int Hout = Hin / 2;
    size_t tot = (size_t)B * Hout * Hout;
    k_downmask<<<(unsigned)((tot + 255) / 256), 256, 0, stream>>>(mi, mo, B, Hin, Hin, Hout, Hout);
  };
  down(m0, m1, 256); down(m1, m2, 128); down(m2, m3, 64); down(m3, m4, 32); down(m4, m5, 16);
  auto maskAt = [&](int H) -> const float* {
    switch (H) { case 128: return m1; case 64: return m2; case 32: return m3; case 16: return m4; default: return m5; }
  };

  // ---- stem ----
  {
    size_t tot = (size_t)B * 128 * 128 * 32;
    k_stem<<<(unsigned)((tot + 255) / 256), 256, 0, stream>>>(
        x, m0, stem_w, stem_bn[0], stem_bn[1], stem_bn[2], stem_bn[3], m1, actA, B);
  }

  auto gemm = [&](const _Float16* A, const _Float16* W,
                  const float* g, const float* bt, const float* mu, const float* var,
                  const float* mask, const _Float16* resid, _Float16* out,
                  int M, int Kp, int Np, int Nreal, int relu6) {
    int waves = (M >> 5) * (Np >> 5);         // 32x32 macro tiles per wave
    int blocks = (waves + 7) / 8;
    k_gemm_bn<<<blocks, 256, 0, stream>>>(A, W, g, bt, mu, var, mask, resid, out,
                                          M, Kp, Np, Nreal, relu6);
  };

  // ---- inverted residual blocks ----
  _Float16* cur = actA;
  _Float16* oth = actB;
  for (int bi = 0; bi < NB; ++bi) {
    const Blk& k = blk[bi];
    int Hin = k.Hin, Hout = (k.stride == 2) ? Hin / 2 : Hin;
    int Min = B * Hin * Hin, Mout = B * Hout * Hout;
    int Kp1 = pad32(k.cin), Hp = pad32(k.hid), Cp = pad32(k.cout);
    const float *b1g = P(bi*15+0), *b1b = P(bi*15+1), *b1m = P(bi*15+2), *b1v = P(bi*15+3);
    const float *b2g = P(bi*15+4), *b2b = P(bi*15+5), *b2m = P(bi*15+6), *b2v = P(bi*15+7);
    const float *b3g = P(bi*15+8), *b3b = P(bi*15+9), *b3m = P(bi*15+10), *b3v = P(bi*15+11);

    // 1x1 expand (WMMA) + BN + ReLU6 * m
    gemm(cur, w1h[bi], b1g, b1b, b1m, b1v, maskAt(Hin), nullptr, oth,
         Min, Kp1, Hp, k.hid, 1);
    // depthwise 3x3 + BN + ReLU6 * m
    {
      size_t tot = (size_t)Mout * Hp;
      int padlo = (k.stride == 2) ? 0 : 1;
      k_dwconv_bn<<<(unsigned)((tot + 255) / 256), 256, 0, stream>>>(
          oth, P(bi * 15 + 13), b2g, b2b, b2m, b2v, maskAt(Hout), actC,
          B, Hin, Hin, Hout, Hout, k.hid, Hp, k.stride, padlo);
    }
    // 1x1 project (WMMA) + BN * m (+ residual)
    bool res = (k.cin == k.cout && k.stride == 1);
    gemm(actC, w3h[bi], b3g, b3b, b3m, b3v, maskAt(Hout), res ? cur : nullptr, oth,
         Mout, Hp, Cp, k.cout, 0);
    _Float16* t = cur; cur = oth; oth = t;
  }

  // ---- head 1x1 (WMMA) + BN + ReLU6 * m5 ----
  gemm(cur, headh, head_bn[0], head_bn[1], head_bn[2], head_bn[3], m5, nullptr, oth,
       B * 64, 320, 1280, 1280, 1);

  // ---- masked global average pool + FC ----
  k_pool<<<(B * 1280 + 255) / 256, 256, 0, stream>>>(oth, m5, pooled, B, 64, 1280);
  k_fc<<<(B * 10 + 255) / 256, 256, 0, stream>>>(pooled, fc_w, fc_b, (float*)d_out, B, 1280, 10);
}
